// GAT4Rec_13142599925974
// MI455X (gfx1250) — matrix operand order repaired and checked by reading, not compiled, not run on hardware
//
#include <hip/hip_runtime.h>
#include <hip/hip_bf16.h>
#include <math.h>

// Problem constants (from reference)
#define BATCH   20000
#define KNBR    32
#define DIM     256
#define HALF    128
#define BTILE   16          // batch rows per workgroup
#define NTHREADS 256        // 8 wave32

typedef float v2f  __attribute__((ext_vector_type(2)));
typedef float v8f  __attribute__((ext_vector_type(8)));
#if !__has_builtin(__builtin_amdgcn_wmma_f32_16x16x4_f32) && \
     __has_builtin(__builtin_amdgcn_wmma_f32_16x16x32_bf16)
typedef __bf16 v16bf __attribute__((ext_vector_type(16)));
#endif

// ---------------------------------------------------------------------------
// Kernel 1: wa1[d] = sum_h W[h][d]*a[h],  wa2[d] = sum_h W[h][d]*a[H+h]
// ---------------------------------------------------------------------------
__global__ void gat_precompute_wa(const float* __restrict__ W,
                                  const float* __restrict__ a,
                                  float* __restrict__ wa) {
    int d = threadIdx.x;                 // 256 threads, 1 block
    float w1 = 0.f, w2 = 0.f;
    for (int h = 0; h < HALF; ++h) {
        float wv = W[h * DIM + d];
        w1 = fmaf(wv, a[h],        w1);
        w2 = fmaf(wv, a[HALF + h], w2);
    }
    wa[d]       = w1;
    wa[DIM + d] = w2;
}

// ---------------------------------------------------------------------------
// Kernel 2: full fused GAT scoring for a 16-row batch tile per workgroup.
// ---------------------------------------------------------------------------
#define XS  260   // xbar LDS row stride (pad vs 64-bank LDS)
#define HS  130   // head/usum LDS row stride

__global__ void __launch_bounds__(NTHREADS)
gat4rec_fused(const int*   __restrict__ u,
              const int*   __restrict__ t_idx,
              const int*   __restrict__ n_idx,
              const float* __restrict__ ent,
              const float* __restrict__ usr,
              const float* __restrict__ Wm,
              const float* __restrict__ wa,
              float*       __restrict__ out) {
    __shared__ float s_wa1[DIM];
    __shared__ float s_wa2[DIM];
    __shared__ int   idx_lds[BTILE * KNBR];
    __shared__ float s_lds[BTILE * KNBR];      // scaled neighbor scores
    __shared__ float scale_lds[BTILE * KNBR];  // norm-clip scales
    __shared__ float wk_lds[BTILE * KNBR];     // alpha * scale
    __shared__ float s0_lds[BTILE];
    __shared__ float usc[BTILE];
    __shared__ float usum_lds[BTILE * HS];     // clipped u[:128]+u[128:]
    __shared__ float xbar_lds[BTILE * XS];     // attention-weighted x
    __shared__ float head_lds[BTILE * HS];

    const int tid  = threadIdx.x;
    const int b0   = blockIdx.x * BTILE;       // 1250 blocks * 16 = 20000

    // --- preload wa1/wa2 ---------------------------------------------------
    s_wa1[tid] = wa[tid];
    s_wa2[tid] = wa[DIM + tid];
    __syncthreads();

    // --- stage 1: per-(b,k) norm + score; xt score; user norms -------------
    for (int p = tid; p < BTILE * KNBR; p += NTHREADS) {
        const int bl = p >> 5, k = p & (KNBR - 1);
        const int idx = n_idx[(size_t)(b0 + bl) * KNBR + k];
        idx_lds[p] = idx;
        const float4* row = (const float4*)(ent + (size_t)idx * DIM);
        float ss = 0.f, dt = 0.f;
        for (int q = 0; q < DIM / 4; ++q) {
            float4 v = row[q];
            ss = fmaf(v.x, v.x, fmaf(v.y, v.y, fmaf(v.z, v.z, fmaf(v.w, v.w, ss))));
            dt = fmaf(v.x, s_wa2[4*q+0], fmaf(v.y, s_wa2[4*q+1],
                 fmaf(v.z, s_wa2[4*q+2], fmaf(v.w, s_wa2[4*q+3], dt))));
        }
        const float scale = fminf(1.f, 1.f / fmaxf(sqrtf(ss), 1e-12f));
        s_lds[p]     = scale * dt;
        scale_lds[p] = scale;
    }
    if (tid < BTILE) {                          // target-entity scores
        const int idx = t_idx[b0 + tid];
        const float4* row = (const float4*)(ent + (size_t)idx * DIM);
        float ss = 0.f, dt = 0.f;
        for (int q = 0; q < DIM / 4; ++q) {
            float4 v = row[q];
            ss = fmaf(v.x, v.x, fmaf(v.y, v.y, fmaf(v.z, v.z, fmaf(v.w, v.w, ss))));
            dt = fmaf(v.x, s_wa1[4*q+0], fmaf(v.y, s_wa1[4*q+1],
                 fmaf(v.z, s_wa1[4*q+2], fmaf(v.w, s_wa1[4*q+3], dt))));
        }
        s0_lds[tid] = fminf(1.f, 1.f / fmaxf(sqrtf(ss), 1e-12f)) * dt;
    } else if (tid < 2 * BTILE) {               // user-row norm scales
        const int b = tid - BTILE;
        const float4* row = (const float4*)(usr + (size_t)u[b0 + b] * DIM);
        float ss = 0.f;
        for (int q = 0; q < DIM / 4; ++q) {
            float4 v = row[q];
            ss = fmaf(v.x, v.x, fmaf(v.y, v.y, fmaf(v.z, v.z, fmaf(v.w, v.w, ss))));
        }
        usc[b] = fminf(1.f, 1.f / fmaxf(sqrtf(ss), 1e-12f));
    }
    __syncthreads();

    // --- stage 2: softmax over K (leaky relu 0.2), fold in clip scale ------
    if (tid < BTILE) {
        const float s0 = s0_lds[tid];
        float v[KNBR];
        float mx = -INFINITY;
        for (int k = 0; k < KNBR; ++k) {
            float t = s0 + s_lds[tid * KNBR + k];
            t = (t >= 0.f) ? t : 0.2f * t;
            v[k] = t;
            mx = fmaxf(mx, t);
        }
        float sum = 0.f;
        for (int k = 0; k < KNBR; ++k) { v[k] = __expf(v[k] - mx); sum += v[k]; }
        const float inv = 1.f / sum;
        for (int k = 0; k < KNBR; ++k)
            wk_lds[tid * KNBR + k] = v[k] * inv * scale_lds[tid * KNBR + k];
    }
    // usum[b][h] = uscale[b] * (u[b][h] + u[b][h+128])
    for (int i = tid; i < BTILE * HALF; i += NTHREADS) {
        const int b = i >> 7, h = i & (HALF - 1);
        const float* ur = usr + (size_t)u[b0 + b] * DIM;
        usum_lds[b * HS + h] = usc[b] * (ur[h] + ur[h + HALF]);
    }
    __syncthreads();

    // --- stage 3: xbar[b][d] = sum_k wk[b][k] * ent[idx[b][k]][d] ----------
    {
        const int d = tid;                      // 256 threads == 256 dims
        for (int b = 0; b < BTILE; ++b) {
            float acc = 0.f;
            const int*   ip = idx_lds + b * KNBR;
            const float* wp = wk_lds  + b * KNBR;
            for (int k = 0; k < KNBR; ++k)
                acc = fmaf(wp[k], ent[(size_t)ip[k] * DIM + d], acc);
            xbar_lds[b * XS + d] = acc;
        }
    }
    __syncthreads();

    // --- stage 4: head = xbar @ W^T via WMMA (each wave: one 16x16 tile) ---
    {
        const int lane = tid & 31;
        const int wv   = tid >> 5;              // wave 0..7 -> h columns
        const int m    = lane & 15;             // A row / C col index
        const int hcol = (wv << 4) + (lane & 15);
        const int kb   = (lane >> 4) << 1;      // f32 K sub-offset: 0 or 2
        v8f c = {};
#if __has_builtin(__builtin_amdgcn_wmma_f32_16x16x4_f32)
        for (int d0 = 0; d0 < DIM; d0 += 4) {
            v2f A, B;
            A.x = xbar_lds[m * XS + d0 + kb];
            A.y = xbar_lds[m * XS + d0 + kb + 1];
            const float* wp = Wm + (size_t)hcol * DIM + d0 + kb;
            B.x = wp[0];
            B.y = wp[1];
            c = __builtin_amdgcn_wmma_f32_16x16x4_f32(
                    false, A, false, B, (short)0, c, false, false);
        }
#elif __has_builtin(__builtin_amdgcn_wmma_f32_16x16x32_bf16)
        for (int d0 = 0; d0 < DIM; d0 += 32) {
            v16bf A, B;
            const int abase = d0 + ((lane >> 4) << 4);   // A half-wave K base
            for (int e = 0; e < 8; ++e) {
                A[e]     = (__bf16)xbar_lds[m * XS + abase + e];
                A[e + 8] = (__bf16)xbar_lds[m * XS + abase + 8 + e];
            }
            const int bbase = d0 + ((lane >> 4) << 4);   // B half-wave K base
            const float* wp = Wm + (size_t)hcol * DIM + bbase;
            for (int e = 0; e < 16; ++e) B[e] = (__bf16)wp[e];
            c = __builtin_amdgcn_wmma_f32_16x16x32_bf16(
                    false, A, false, B, (short)0, c, false, false);
        }
#else
        for (int j = 0; j < 8; ++j) {
            const int mr = j + ((lane >> 4) << 3);
            float acc = 0.f;
            for (int d0 = 0; d0 < DIM; ++d0)
                acc = fmaf(xbar_lds[mr * XS + d0], Wm[(size_t)hcol * DIM + d0], acc);
            c[j] = acc;
        }
#endif
        // C tile layout: VGPR j -> row m=j (lanes 0-15) / m=j+8 (lanes 16-31)
        for (int j = 0; j < 8; ++j) {
            const int mr = j + ((lane >> 4) << 3);
            head_lds[mr * HS + hcol] = c[j];
        }
    }
    __syncthreads();

    // --- stage 5: uv = head . usum ; out = sigmoid(uv) ---------------------
    if (tid < BTILE) {
        float acc = 0.f;
        for (int h = 0; h < HALF; ++h)
            acc = fmaf(head_lds[tid * HS + h], usum_lds[tid * HS + h], acc);
        out[b0 + tid] = 1.f / (1.f + __expf(-acc));
    }
}

// ---------------------------------------------------------------------------
extern "C" void kernel_launch(void* const* d_in, const int* in_sizes, int n_in,
                              void* d_out, int out_size, void* d_ws, size_t ws_size,
                              hipStream_t stream) {
    const int*   u     = (const int*)  d_in[0];   // [B]
    const int*   t_idx = (const int*)  d_in[1];   // [B]
    const int*   n_idx = (const int*)  d_in[2];   // [B,K]
    const float* ent   = (const float*)d_in[3];   // [N_ENT, D]
    const float* usr   = (const float*)d_in[4];   // [N_USR, D]
    const float* Wm    = (const float*)d_in[5];   // [H, D]
    const float* a     = (const float*)d_in[6];   // [D]
    float*       out   = (float*)d_out;           // [B]
    float*       wa    = (float*)d_ws;            // 512 floats scratch

    gat_precompute_wa<<<1, DIM, 0, stream>>>(Wm, a, wa);
    gat4rec_fused<<<BATCH / BTILE, NTHREADS, 0, stream>>>(
        u, t_idx, n_idx, ent, usr, Wm, wa, out);
}
